// EnhancementGenerator_919123001585
// MI455X (gfx1250) — matrix operand order repaired and checked by reading, not compiled, not run on hardware
//
#include <hip/hip_runtime.h>
#include <math.h>

typedef _Float16 v16h __attribute__((ext_vector_type(16)));
typedef _Float16 v8h  __attribute__((ext_vector_type(8)));
typedef float    v8f  __attribute__((ext_vector_type(8)));

#define BB   64
#define TT   600
#define INF  257
#define NN   (BB * TT)      // 38400 rows, divisible by 32
#define HH   40
#define G3   120            // 3*H
#define KP1  288            // 257 padded to 9 k-chunks of 32
#define NP1  128            // 120 padded to 8 n-tiles (even)
#define KIN_KAN 720         // 80 * 9 (silu + 8 spline bases)
#define KPA  736            // 720 padded to 23 k-chunks of 32
#define O1   80
#define O2   257
#define NPA1 96             // 80 padded to 6 n-tiles (even, for 2N blocking)
#define NP2  288            // 257 padded to 18 n-tiles (even)

__device__ __forceinline__ float sigmoidf_(float x) { return 1.0f / (1.0f + __expf(-x)); }

// ---------------- packing kernels ----------------

__global__ __launch_bounds__(256) void pack_x_kernel(const float* __restrict__ x,
                                                     _Float16* __restrict__ xh) {
    int idx = blockIdx.x * 256 + threadIdx.x;
    if (idx >= NN * KP1) return;
    int n = idx / KP1, k = idx % KP1;
    xh[idx] = (k < INF) ? (_Float16)x[(size_t)n * INF + k] : (_Float16)0.f;
}

__global__ __launch_bounds__(256) void pack_w_kernel(const float* __restrict__ W,
                                                     _Float16* __restrict__ Wh,
                                                     int Nout, int Kin, int NPAD, int KPAD) {
    int idx = blockIdx.x * 256 + threadIdx.x;
    if (idx >= NPAD * KPAD) return;
    int r = idx / KPAD, c = idx % KPAD;
    Wh[idx] = (r < Nout && c < Kin) ? (_Float16)W[(size_t)r * Kin + c] : (_Float16)0.f;
}

// Wa = [base_w | (spline_w * scaler).reshape(out, 640)] padded to (NPAD x 736)
__global__ __launch_bounds__(256) void pack_wa_kernel(const float* __restrict__ base,
                                                      const float* __restrict__ spline,
                                                      const float* __restrict__ scaler,
                                                      _Float16* __restrict__ Wh,
                                                      int Nout, int NPAD) {
    int idx = blockIdx.x * 256 + threadIdx.x;
    if (idx >= NPAD * KPA) return;
    int o = idx / KPA, c = idx % KPA;
    float v = 0.f;
    if (o < Nout) {
        if (c < O1) {
            v = base[(size_t)o * O1 + c];
        } else if (c < KIN_KAN) {
            int i = (c - O1) >> 3, g = (c - O1) & 7;
            v = spline[((size_t)o * O1 + i) * 8 + g] * scaler[(size_t)o * O1 + i];
        }
    }
    Wh[idx] = (_Float16)v;
}

// ---------------- KAN feature builder: F = [silu(x) | bspline bases] ----------------

__global__ __launch_bounds__(256) void kan_feat_kernel(const float* __restrict__ act,
                                                       _Float16* __restrict__ feat) {
    int idx = blockIdx.x * 256 + threadIdx.x;
    int n = idx / 96, i = idx % 96;
    if (n >= NN) return;
    _Float16* frow = feat + (size_t)n * KPA;
    if (i < O1) {
        float x = act[(size_t)n * O1 + i];
        frow[i] = (_Float16)(x * sigmoidf_(x));
        // Cox-de Boor, uniform grid g_j = -2.2 + 0.4j (j=0..11), order 3 -> 8 bases
        float b[11];
        #pragma unroll
        for (int j = 0; j < 11; ++j) {
            float gj = -2.2f + 0.4f * (float)j;
            b[j] = (x >= gj && x < gj + 0.4f) ? 1.0f : 0.0f;
        }
        #pragma unroll
        for (int k = 1; k <= 3; ++k) {
            for (int j = 0; j < 11 - k; ++j) {
                float gj  = -2.2f + 0.4f * (float)j;
                float inv = 1.0f / (0.4f * (float)k);
                float left  = (x - gj) * inv;
                float right = ((gj + 0.4f * (float)(k + 1)) - x) * inv;
                b[j] = left * b[j] + right * b[j + 1];
            }
        }
        #pragma unroll
        for (int g = 0; g < 8; ++g) frow[O1 + i * 8 + g] = (_Float16)b[g];
    } else {
        frow[KIN_KAN + (i - O1)] = (_Float16)0.f;   // zero K padding (720..735)
    }
}

// ---------------- WMMA GEMM: C(MxNcols,f32) = A(MxK,f16) * Bw(NxK,f16)^T ----------------
// Per wave: 2 M-tiles x 2 N-tiles (4 accumulators): 8x b128 loads feed 4 WMMAs per k-chunk
// (2 loads/WMMA). NPAD rows of Bw are zero-padded so the k-loop needs no per-lane guards
// (EXEC stays all-1s for WMMA); stores are column-guarded.
// MODE 0: += bias[col]   MODE 1: plain   MODE 2: 1.2*sigmoid(slope[col]*v)

template <int MODE>
__global__ __launch_bounds__(256) void wmma_gemm_kernel(
    const _Float16* __restrict__ A, const _Float16* __restrict__ Bw,
    float* __restrict__ C, int K, int ntiles_n, int Ncols, int total_waves,
    const float* __restrict__ bias, const float* __restrict__ slope) {
    const int lane = threadIdx.x & 31;
    const int gt = blockIdx.x * 8 + (threadIdx.x >> 5);
    if (gt >= total_waves) return;                  // wave-uniform
    const int ntn2 = ntiles_n >> 1;
    const int tm = (gt / ntn2) * 2;
    const int tn = (gt % ntn2) * 2;
    const int hs = lane >> 4;                       // half-wave select
    const int lh = lane & 15;

    const _Float16* a0p = A + (size_t)(tm * 16 + lh) * K;
    const _Float16* a1p = a0p + (size_t)16 * K;
    const _Float16* b0p = Bw + (size_t)(tn * 16 + lh) * K;
    const _Float16* b1p = b0p + (size_t)16 * K;
    const int kA = hs * 8;                          // A: K 0-7 / 8-15 (+16 for hi half)
    const int kB = hs * 8;                          // B: K 0-7 / 8-15 (+16 for hi half)

    v8f acc00 = {}; v8f acc01 = {}; v8f acc10 = {}; v8f acc11 = {};
    for (int kc = 0; kc < K; kc += 32) {
        v8h al0 = *(const v8h*)(a0p + kc + kA);
        v8h ah0 = *(const v8h*)(a0p + kc + kA + 16);
        v8h al1 = *(const v8h*)(a1p + kc + kA);
        v8h ah1 = *(const v8h*)(a1p + kc + kA + 16);
        v8h bl0 = *(const v8h*)(b0p + kc + kB);
        v8h bh0 = *(const v8h*)(b0p + kc + kB + 16);
        v8h bl1 = *(const v8h*)(b1p + kc + kB);
        v8h bh1 = *(const v8h*)(b1p + kc + kB + 16);
        __builtin_prefetch(a0p + kc + 128, 0, 0);   // global_prefetch_b8
        __builtin_prefetch(a1p + kc + 128, 0, 0);
        v16h a0, a1, b0, b1;
        #pragma unroll
        for (int e = 0; e < 8; ++e) {
            a0[e] = al0[e]; a0[8 + e] = ah0[e];
            a1[e] = al1[e]; a1[8 + e] = ah1[e];
            b0[e] = bl0[e]; b0[8 + e] = bh0[e];
            b1[e] = bl1[e]; b1[8 + e] = bh1[e];
        }
        acc00 = __builtin_amdgcn_wmma_f32_16x16x32_f16(false, a0, false, b0, (short)0, acc00, false, false);
        acc01 = __builtin_amdgcn_wmma_f32_16x16x32_f16(false, a0, false, b1, (short)0, acc01, false, false);
        acc10 = __builtin_amdgcn_wmma_f32_16x16x32_f16(false, a1, false, b0, (short)0, acc10, false, false);
        acc11 = __builtin_amdgcn_wmma_f32_16x16x32_f16(false, a1, false, b1, (short)0, acc11, false, false);
    }

    const int col0 = tn * 16 + lh;
    const int col1 = col0 + 16;
    const bool ok0 = col0 < Ncols;
    const bool ok1 = col1 < Ncols;
    float badd0 = (MODE == 0 && ok0) ? bias[col0] : 0.f;
    float badd1 = (MODE == 0 && ok1) ? bias[col1] : 0.f;
    float sl0   = (MODE == 2 && ok0) ? slope[col0] : 0.f;
    float sl1   = (MODE == 2 && ok1) ? slope[col1] : 0.f;
    #pragma unroll
    for (int r = 0; r < 8; ++r) {
        int row0 = tm * 16 + r + hs * 8;            // C/D layout: lanes 16-31 hold M=8+r
        int row1 = row0 + 16;
        float v00 = acc00[r], v01 = acc01[r], v10 = acc10[r], v11 = acc11[r];
        if (MODE == 0) { v00 += badd0; v01 += badd1; v10 += badd0; v11 += badd1; }
        if (MODE == 2) {
            v00 = 1.2f * sigmoidf_(sl0 * v00); v01 = 1.2f * sigmoidf_(sl1 * v01);
            v10 = 1.2f * sigmoidf_(sl0 * v10); v11 = 1.2f * sigmoidf_(sl1 * v11);
        }
        if (ok0) {
            C[(size_t)row0 * Ncols + col0] = v00;
            C[(size_t)row1 * Ncols + col0] = v10;
        }
        if (ok1) {
            C[(size_t)row0 * Ncols + col1] = v01;
            C[(size_t)row1 * Ncols + col1] = v11;
        }
    }
}

// ---------------- GRU scan (sequential over T, parallel over batch x direction) ----------------
// gi0 (x @ Wih0^T + bih0) precomputed by WMMA GEMM; per step only 40x120 mat-vecs remain.

__global__ __launch_bounds__(256) void gru_scan_kernel(
    const float* __restrict__ gi0f, const float* __restrict__ gi0b,
    const float* __restrict__ f0Whh, const float* __restrict__ f0bhh,
    const float* __restrict__ f1Wih, const float* __restrict__ f1bih,
    const float* __restrict__ f1Whh, const float* __restrict__ f1bhh,
    const float* __restrict__ b0Whh, const float* __restrict__ b0bhh,
    const float* __restrict__ b1Wih, const float* __restrict__ b1bih,
    const float* __restrict__ b1Whh, const float* __restrict__ b1bhh,
    float* __restrict__ conc) {
    __shared__ float sW0[G3 * HH], sW1i[G3 * HH], sW1h[G3 * HH];
    __shared__ float sb0[G3], sb1i[G3], sb1h[G3];
    __shared__ float h0s[2][HH], h1s[2][HH];
    __shared__ float g0s[2][G3], g1is[2][G3], g1hs[2][G3];

    const int tid = threadIdx.x;
    const int dir = blockIdx.x & 1;
    const int b0r = (blockIdx.x >> 1) * 2;

    const float* Whh0 = dir ? b0Whh : f0Whh;
    const float* bhh0 = dir ? b0bhh : f0bhh;
    const float* Wih1 = dir ? b1Wih : f1Wih;
    const float* bih1 = dir ? b1bih : f1bih;
    const float* Whh1 = dir ? b1Whh : f1Whh;
    const float* bhh1 = dir ? b1bhh : f1bhh;
    const float* gi0  = dir ? gi0b  : gi0f;

    for (int i = tid; i < G3 * HH; i += 256) { sW0[i] = Whh0[i]; sW1i[i] = Wih1[i]; sW1h[i] = Whh1[i]; }
    for (int i = tid; i < G3; i += 256)      { sb0[i] = bhh0[i]; sb1i[i] = bih1[i]; sb1h[i] = bhh1[i]; }
    if (tid < 2 * HH) { ((float*)h0s)[tid] = 0.f; ((float*)h1s)[tid] = 0.f; }
    __syncthreads();

    for (int t = 0; t < TT; ++t) {
        // gh0 = h0 @ Whh0^T + bhh0
        if (tid < 2 * G3) {
            int r = tid / G3, o = tid % G3;
            float acc = sb0[o];
            const float* w = &sW0[o * HH];
            #pragma unroll
            for (int k = 0; k < HH; ++k) acc += h0s[r][k] * w[k];
            g0s[r][o] = acc;
        }
        __syncthreads();
        // layer0 gates (gi0 rows indexed t for fwd, T-1-t for bwd)
        if (tid < 2 * HH) {
            int r = tid / HH, j = tid % HH;
            int b = b0r + r;
            int tt = dir ? (TT - 1 - t) : t;
            const float* gi = gi0 + ((size_t)b * TT + tt) * (size_t)G3;
            float rr = sigmoidf_(gi[j] + g0s[r][j]);
            float z  = sigmoidf_(gi[HH + j] + g0s[r][HH + j]);
            float nn = tanhf(gi[2 * HH + j] + rr * g0s[r][2 * HH + j]);
            float hp = h0s[r][j];
            h0s[r][j] = (1.0f - z) * nn + z * hp;
        }
        __syncthreads();
        // gi1 = h0 @ Wih1^T + bih1 ; gh1 = h1 @ Whh1^T + bhh1
        for (int d = tid; d < 2 * 2 * G3; d += 256) {
            int r = d / (2 * G3);
            int q = d % (2 * G3);
            int mat = q / G3, o = q % G3;
            float acc; const float* w; const float* hv;
            if (mat == 0) { acc = sb1i[o]; w = &sW1i[o * HH]; hv = h0s[r]; }
            else          { acc = sb1h[o]; w = &sW1h[o * HH]; hv = h1s[r]; }
            #pragma unroll
            for (int k = 0; k < HH; ++k) acc += hv[k] * w[k];
            if (mat == 0) g1is[r][o] = acc; else g1hs[r][o] = acc;
        }
        __syncthreads();
        // layer1 gates + write concat output (hb stored at step index t, no flip)
        if (tid < 2 * HH) {
            int r = tid / HH, j = tid % HH;
            int b = b0r + r;
            float rr = sigmoidf_(g1is[r][j] + g1hs[r][j]);
            float z  = sigmoidf_(g1is[r][HH + j] + g1hs[r][HH + j]);
            float nn = tanhf(g1is[r][2 * HH + j] + rr * g1hs[r][2 * HH + j]);
            float hp = h1s[r][j];
            float hnew = (1.0f - z) * nn + z * hp;
            h1s[r][j] = hnew;
            conc[((size_t)b * TT + t) * (size_t)(2 * HH) + dir * HH + j] = hnew;
        }
        __syncthreads();
    }
}

// ---------------- launcher ----------------

extern "C" void kernel_launch(void* const* d_in, const int* in_sizes, int n_in,
                              void* d_out, int out_size, void* d_ws, size_t ws_size,
                              hipStream_t stream) {
    (void)in_sizes; (void)n_in; (void)out_size; (void)ws_size;
    const float* x       = (const float*)d_in[0];
    // d_in[1] = lengths (unused by reference forward)
    const float* f0_Wih  = (const float*)d_in[2];
    const float* f0_Whh  = (const float*)d_in[3];
    const float* f0_bih  = (const float*)d_in[4];
    const float* f0_bhh  = (const float*)d_in[5];
    const float* f1_Wih  = (const float*)d_in[6];
    const float* f1_Whh  = (const float*)d_in[7];
    const float* f1_bih  = (const float*)d_in[8];
    const float* f1_bhh  = (const float*)d_in[9];
    const float* b0_Wih  = (const float*)d_in[10];
    const float* b0_Whh  = (const float*)d_in[11];
    const float* b0_bih  = (const float*)d_in[12];
    const float* b0_bhh  = (const float*)d_in[13];
    const float* b1_Wih  = (const float*)d_in[14];
    const float* b1_Whh  = (const float*)d_in[15];
    const float* b1_bih  = (const float*)d_in[16];
    const float* b1_bhh  = (const float*)d_in[17];
    const float* k1_base   = (const float*)d_in[18];
    const float* k1_spline = (const float*)d_in[19];
    const float* k1_scaler = (const float*)d_in[20];
    const float* k2_base   = (const float*)d_in[21];
    const float* k2_spline = (const float*)d_in[22];
    const float* k2_scaler = (const float*)d_in[23];
    const float* slope     = (const float*)d_in[24];
    float* out = (float*)d_out;

    // Workspace layout (phase-overlapped; ~84.3 MB total, 16B-aligned offsets):
    char* ws = (char*)d_ws;
    _Float16* xh   = (_Float16*)(ws + 0);          // 38400*288*2 = 22,118,400 (phase A)
    float*    gi0f = (float*)(ws + 22118400);      // 38400*120*4 = 18,432,000 (A,B)
    float*    gi0b = (float*)(ws + 40550400);      // 18,432,000               (A,B)
    _Float16* feat = (_Float16*)(ws + 0);          // 38400*736*2 = 56,524,800 (C,D; reuses xh/gi0)
    float*    conc = (float*)(ws + 58982400);      // 38400*80*4  = 12,288,000 (B->C)
    float*    h1o  = (float*)(ws + 71270400);      // 12,288,000               (C->D)
    _Float16* w0f  = (_Float16*)(ws + 83558400);   // 128*288*2 =  73,728
    _Float16* w0b  = (_Float16*)(ws + 83632128);   // 128*288*2 =  73,728
    _Float16* wa1  = (_Float16*)(ws + 83705856);   //  96*736*2 = 141,312
    _Float16* wa2  = (_Float16*)(ws + 83847168);   // 288*736*2 = 423,936 (ends 84,271,104)

    dim3 blk(256);

    // Phase A: pack inputs/weights, GRU layer-0 input GEMMs (WMMA)
    pack_x_kernel<<<(NN * KP1) / 256, blk, 0, stream>>>(x, xh);
    pack_w_kernel<<<(NP1 * KP1 + 255) / 256, blk, 0, stream>>>(f0_Wih, w0f, G3, INF, NP1, KP1);
    pack_w_kernel<<<(NP1 * KP1 + 255) / 256, blk, 0, stream>>>(b0_Wih, w0b, G3, INF, NP1, KP1);
    pack_wa_kernel<<<(NPA1 * KPA + 255) / 256, blk, 0, stream>>>(k1_base, k1_spline, k1_scaler, wa1, O1, NPA1);
    pack_wa_kernel<<<(NP2 * KPA + 255) / 256, blk, 0, stream>>>(k2_base, k2_spline, k2_scaler, wa2, O2, NP2);

    const int tw1 = (NN / 32) * (NP1 / 32);   // 1200 * 4 = 4800 waves
    wmma_gemm_kernel<0><<<tw1 / 8, blk, 0, stream>>>(xh, w0f, gi0f, KP1, NP1 / 16, G3, tw1, f0_bih, nullptr);
    wmma_gemm_kernel<0><<<tw1 / 8, blk, 0, stream>>>(xh, w0b, gi0b, KP1, NP1 / 16, G3, tw1, b0_bih, nullptr);

    // Phase B: sequential bidirectional GRU scan (batch/direction parallel)
    gru_scan_kernel<<<64, blk, 0, stream>>>(gi0f, gi0b,
        f0_Whh, f0_bhh, f1_Wih, f1_bih, f1_Whh, f1_bhh,
        b0_Whh, b0_bhh, b1_Wih, b1_bih, b1_Whh, b1_bhh, conc);

    // Phase C: KAN1 = [silu|splines](conc) @ wa1^T  (WMMA, f32 out)
    kan_feat_kernel<<<(NN * 96) / 256, blk, 0, stream>>>(conc, feat);
    const int tw2 = (NN / 32) * (NPA1 / 32);  // 1200 * 3 = 3600 waves
    wmma_gemm_kernel<1><<<tw2 / 8, blk, 0, stream>>>(feat, wa1, h1o, KPA, NPA1 / 16, O1, tw2, nullptr, nullptr);

    // Phase D: KAN2 + fused 1.2*sigmoid(slope*x) epilogue -> d_out
    kan_feat_kernel<<<(NN * 96) / 256, blk, 0, stream>>>(h1o, feat);
    const int tw3 = (NN / 32) * (NP2 / 32);   // 1200 * 9 = 10800 waves
    wmma_gemm_kernel<2><<<tw3 / 8, blk, 0, stream>>>(feat, wa2, out, KPA, NP2 / 16, O2, tw3, nullptr, slope);
}
